// QLoRABigNet_90271622627687
// MI455X (gfx1250) — matrix operand order repaired and checked by reading, not compiled, not run on hardware
//
#include <hip/hip_runtime.h>
#include <hip/hip_bf16.h>

// ---------------------------------------------------------------------------
// QLoRA BigNet forward on MI455X (gfx1250, wave32, WMMA + TDM).
// Heavy math: v_wmma_f32_16x16x32_bf16. Tile feed: tensor_load_to_lds (TDM)
// double-buffered into LDS with descriptor padding (80B row stride) to kill
// bank conflicts; fragments come from LDS via ds_load_b128.
// Weights dequantized once (int4 group-quant -> bf16) into d_ws; W (2MB/layer)
// and the bf16 activation ping-pong stay L2-resident (192 MB L2).
// ---------------------------------------------------------------------------

typedef __attribute__((ext_vector_type(16))) __bf16 v16bf;
typedef __attribute__((ext_vector_type(8)))  __bf16 v8bf;
typedef __attribute__((ext_vector_type(8)))  float  v8f;
typedef unsigned int u32x4 __attribute__((ext_vector_type(4)));
typedef int          i32x8 __attribute__((ext_vector_type(8)));
typedef int          i32x4 __attribute__((ext_vector_type(4)));

#define DIM    1024
#define BATCH  16384
#define NLAY   18
#define RANK   32
#define KSTEP  32
#define LDS_STRIDE 40   // 32 bf16 data + 8 bf16 (16B) TDM pad -> 80B row stride

// ---- WMMA fragment loaders (contiguous K order, same in A and B) -----
// 16-bit A 16x32 (MxK), this code's contiguous K: lanes 0-15 hold M=lane, K in {0..15};
// lanes 16-31 hold M=lane-16, K in {16..31}.
__device__ __forceinline__ v16bf load_a16(const __bf16* __restrict__ base,
                                          int stride, int mbase, int lane, int k) {
  const int m  = lane & 15;
  const int ak = (lane >> 4) << 4;          // 0 or 16
  const __bf16* p = base + (size_t)(mbase + m) * stride + k + ak;
  v8bf lo = *(const v8bf*)p;                // K = ak .. ak+7
  v8bf hi = *(const v8bf*)(p + 8);          // K = ak+8 .. ak+15
  v16bf r;
#pragma unroll
  for (int i = 0; i < 8; ++i) { r[i] = lo[i]; r[i + 8] = hi[i]; }
  return r;
}

// 16-bit B 32x16 (KxN), same contiguous K as A: lane n (0-15) holds K=0..15 of column n,
// lane n+16 holds K=16..31. Column n = row (obase+n) of row-major weights.
__device__ __forceinline__ v16bf load_b16(const __bf16* __restrict__ base,
                                          int stride, int obase, int lane, int k) {
  const int n  = lane & 15;
  const int bk = (lane >> 4) << 4;          // 0 or 16
  return *(const v16bf*)(base + (size_t)(obase + n) * stride + k + bk);
}

__device__ __forceinline__ v8f wmma_bf16(v16bf a, v16bf b, v8f c) {
  return __builtin_amdgcn_wmma_f32_16x16x32_bf16(false, a, false, b,
                                                 (short)0, c, false, false);
}

// ---- TDM: LDS byte offset of a __shared__ object --------------------------
typedef __attribute__((address_space(3))) const void* lds_cptr;
__device__ __forceinline__ unsigned lds_off(const void* p) {
  return (unsigned)(unsigned long long)(lds_cptr)p;
}

// ---- TDM: 2D tile load, 128 rows x 32 bf16 --------------------------------
// Descriptor layout per CDNA5 ISA.
// Pad: 4 dwords after every 16 dwords -> LDS row stride 80B (bank-conflict free
// for the 16-lane, 80B-strided fragment reads: gcd(20,64)=4 -> 16 banks).
__device__ __forceinline__ void tdm_tile_128x32(const __bf16* gsrc,
                                                unsigned ldsoff) {
  const unsigned long long ga = (unsigned long long)gsrc;
  u32x4 g0;
  g0[0] = 1u;                                    // count=1 (valid), user mode
  g0[1] = ldsoff;                                // lds_addr (bytes)
  g0[2] = (unsigned)ga;                          // global_addr[31:0]
  g0[3] = (unsigned)((ga >> 32) & 0x01ffffffu)   // global_addr[56:32]
        | (2u << 30);                            // type = 2 ("image")
  i32x8 g1;
  g1[0] = (1 << 16)                              // data_size = 1 -> 2 bytes
        | (1 << 20)                              // pad_enable
        | (3 << 22)                              // pad_interval: 16 dwords
        | (3 << 25);                             // pad_amount: 4 dwords
  g1[1] = (int)((1024u & 0xffffu) << 16);        // tensor_dim0[15:0] = 1024
  g1[2] = (int)((1024u >> 16)                    // tensor_dim0[31:16]
        | (128u << 16));                         // tensor_dim1[15:0] = 128
  g1[3] = (int)(32u << 16);                      // tensor_dim1 hi=0; tile_dim0=32
  g1[4] = 128;                                   // tile_dim1=128, tile_dim2=0
  g1[5] = 1024;                                  // tensor_dim0_stride[31:0]
  g1[6] = 0;                                     // stride hi / dim1_stride lo
  g1[7] = 0;
  i32x4 z4 = {0, 0, 0, 0};
  i32x8 z8 = {0, 0, 0, 0, 0, 0, 0, 0};
  // clang-23 / therock-10.0 form: 6 args (g0, g1, g2, g3, extra, cpol)
  __builtin_amdgcn_tensor_load_to_lds(g0, g1, z4, z4, z8, 0);
}

// ---- prep kernels ----------------------------------------------------------

__global__ __launch_bounds__(256) void dequant_w_kernel(
    const int* __restrict__ qw, const float* __restrict__ scale,
    __bf16* __restrict__ W) {
  const unsigned g = blockIdx.x * 256u + threadIdx.x;   // 18*1024*64 groups
  if (g >= (unsigned)(NLAY * DIM * 64)) return;
  const int* q = qw + (size_t)g * 16;
  const float s = scale[g];
  __bf16* w = W + (size_t)g * 16;
#pragma unroll
  for (int i = 0; i < 16; ++i)
    w[i] = (__bf16)(((float)q[i] * (2.0f / 15.0f) - 1.0f) * s);
}

__global__ __launch_bounds__(256) void f32_to_bf16_kernel(
    const float* __restrict__ in, __bf16* __restrict__ out, int n) {
  const int i = blockIdx.x * 256 + threadIdx.x;
  if (i < n) out[i] = (__bf16)in[i];
}

// ---- t = x @ A^T  (16384 x 32), one wave per 16x32 output tile -------------
__global__ __launch_bounds__(256) void lora_t_kernel(
    const __bf16* __restrict__ X, const __bf16* __restrict__ A,
    __bf16* __restrict__ T) {
  const int lane  = threadIdx.x & 31;
  const int wave  = threadIdx.x >> 5;
  const int mbase = blockIdx.x * 128 + wave * 16;
  v8f acc[2] = {};
  for (int k = 0; k < DIM; k += KSTEP) {
    v16bf a = load_a16(X, DIM, mbase, lane, k);
#pragma unroll
    for (int j = 0; j < 2; ++j) {
      v16bf b = load_b16(A, DIM, j * 16, lane, k);
      acc[j] = wmma_bf16(a, b, acc[j]);
    }
  }
  const int n  = lane & 15;
  const int mo = (lane >> 4) * 8;
#pragma unroll
  for (int j = 0; j < 2; ++j)
#pragma unroll
    for (int r = 0; r < 8; ++r)
      T[(size_t)(mbase + mo + r) * RANK + j * 16 + n] = (__bf16)acc[j][r];
}

// ---- main fused layer: Y = relu?(X @ W^T + bias + T @ LB^T) ----------------
// Block = 256 threads = 8 waves (4 row-groups x 2 col-groups).
// Wave tile = 32 rows x 64 cols = 2x4 WMMA accumulators.
// Tiles (128x32) TDM-DMA'd into double-buffered LDS; wave 0 drives the TDM.
__global__ __launch_bounds__(256) void qlora_gemm_kernel(
    const __bf16* __restrict__ X, const __bf16* __restrict__ W,
    const float* __restrict__ bias, const __bf16* __restrict__ T,
    const __bf16* __restrict__ LB, __bf16* __restrict__ Y, int relu) {
  __shared__ __bf16 Xt[2][128 * LDS_STRIDE];   // 2 x 10 KB
  __shared__ __bf16 Wt[2][128 * LDS_STRIDE];   // 2 x 10 KB

  const int lane = threadIdx.x & 31;
  const int wave = threadIdx.x >> 5;
  const int wr   = wave & 3;
  const int wc   = wave >> 2;
  const int mb   = blockIdx.x * 128;           // block row base
  const int ob   = blockIdx.y * 128;           // block col base

  // Preload K-step 0.
  if (wave == 0) {
    tdm_tile_128x32(X + (size_t)mb * DIM, lds_off(&Xt[0][0]));
    tdm_tile_128x32(W + (size_t)ob * DIM, lds_off(&Wt[0][0]));
    __builtin_amdgcn_s_wait_tensorcnt(0);
  }
  __syncthreads();

  v8f acc[2][4] = {};
  for (int ks = 0; ks < DIM / KSTEP; ++ks) {
    const int cur = ks & 1;
    if (wave == 0 && ks + 1 < DIM / KSTEP) {   // prefetch next tiles via TDM
      const int kn = (ks + 1) * KSTEP;
      tdm_tile_128x32(X + (size_t)mb * DIM + kn, lds_off(&Xt[cur ^ 1][0]));
      tdm_tile_128x32(W + (size_t)ob * DIM + kn, lds_off(&Wt[cur ^ 1][0]));
    }
    const __bf16* xs = &Xt[cur][0];
    const __bf16* ws = &Wt[cur][0];
    v16bf a[2], b[4];
#pragma unroll
    for (int i = 0; i < 2; ++i)
      a[i] = load_a16(xs, LDS_STRIDE, wr * 32 + i * 16, lane, 0);
#pragma unroll
    for (int j = 0; j < 4; ++j)
      b[j] = load_b16(ws, LDS_STRIDE, wc * 64 + j * 16, lane, 0);
#pragma unroll
    for (int i = 0; i < 2; ++i)
#pragma unroll
      for (int j = 0; j < 4; ++j)
        acc[i][j] = wmma_bf16(a[i], b[j], acc[i][j]);
    if (wave == 0) __builtin_amdgcn_s_wait_tensorcnt(0);
    __syncthreads();   // next-buffer DMA complete + all reads of cur done
  }

  const int mbase = mb + wr * 32;
  const int obase = ob + wc * 64;

  // LoRA contribution: one extra K=32 WMMA per tile (D += t(16x32) @ lb(32x16))
  v16bf ta[2], lv[4];
#pragma unroll
  for (int i = 0; i < 2; ++i) ta[i] = load_a16(T, RANK, mbase + i * 16, lane, 0);
#pragma unroll
  for (int j = 0; j < 4; ++j) lv[j] = load_b16(LB, RANK, obase + j * 16, lane, 0);
#pragma unroll
  for (int i = 0; i < 2; ++i)
#pragma unroll
    for (int j = 0; j < 4; ++j)
      acc[i][j] = wmma_bf16(ta[i], lv[j], acc[i][j]);

  // Epilogue: D layout = VGPR r holds M=r (lanes 0-15) / M=r+8 (lanes 16-31),
  // N = lane & 15.
  const int n  = lane & 15;
  const int mo = (lane >> 4) * 8;
#pragma unroll
  for (int j = 0; j < 4; ++j) {
    const float bv = bias[obase + j * 16 + n];
#pragma unroll
    for (int i = 0; i < 2; ++i) {
#pragma unroll
      for (int r = 0; r < 8; ++r) {
        float v = acc[i][j][r] + bv;
        if (relu) v = fmaxf(v, 0.0f);
        Y[(size_t)(mbase + i * 16 + mo + r) * DIM + obase + j * 16 + n] = (__bf16)v;
      }
    }
  }
}

// ---- residual + LayerNorm (blocks 0..4): out = LN(y + r) -> bf16 -----------
__global__ __launch_bounds__(256) void resid_ln_kernel(
    const __bf16* __restrict__ Y, const __bf16* __restrict__ R,
    const float* __restrict__ g, const float* __restrict__ beta,
    __bf16* __restrict__ Out) {
  const int row = blockIdx.x;
  const int tid = threadIdx.x;
  __shared__ float s_sum[8], s_ss[8];
  float v[4];
  float sum = 0.f, ss = 0.f;
  const __bf16* y = Y + (size_t)row * DIM;
  const __bf16* r = R + (size_t)row * DIM;
#pragma unroll
  for (int j = 0; j < 4; ++j) {
    const int c = tid + j * 256;
    const float t = (float)y[c] + (float)r[c];
    v[j] = t; sum += t; ss += t * t;
  }
#pragma unroll
  for (int off = 16; off > 0; off >>= 1) {
    sum += __shfl_xor(sum, off, 32);
    ss  += __shfl_xor(ss,  off, 32);
  }
  const int wv = tid >> 5, lane = tid & 31;
  if (lane == 0) { s_sum[wv] = sum; s_ss[wv] = ss; }
  __syncthreads();
  if (tid == 0) {
    float a = 0.f, b = 0.f;
#pragma unroll
    for (int i = 0; i < 8; ++i) { a += s_sum[i]; b += s_ss[i]; }
    s_sum[0] = a; s_ss[0] = b;
  }
  __syncthreads();
  const float mean = s_sum[0] * (1.0f / DIM);
  const float var  = s_ss[0]  * (1.0f / DIM) - mean * mean;
  const float rs   = rsqrtf(var + 1e-5f);
#pragma unroll
  for (int j = 0; j < 4; ++j) {
    const int c = tid + j * 256;
    Out[(size_t)row * DIM + c] = (__bf16)((v[j] - mean) * rs * g[c] + beta[c]);
  }
}

// ---- final residual (block 5, no LN): fp32 out -----------------------------
__global__ __launch_bounds__(256) void resid_out_kernel(
    const __bf16* __restrict__ Y, const __bf16* __restrict__ R,
    float* __restrict__ Out) {
  const size_t i = (size_t)blockIdx.x * 256 + threadIdx.x;
  Out[i] = (float)Y[i] + (float)R[i];
}

// ---------------------------------------------------------------------------

extern "C" void kernel_launch(void* const* d_in, const int* in_sizes, int n_in,
                              void* d_out, int out_size, void* d_ws, size_t ws_size,
                              hipStream_t stream) {
  const float* x     = (const float*)d_in[0];
  const float* scale = (const float*)d_in[1];
  const float* bias  = (const float*)d_in[2];
  const float* la    = (const float*)d_in[3];
  const float* lb    = (const float*)d_in[4];
  const float* ln_w  = (const float*)d_in[5];
  const float* ln_b  = (const float*)d_in[6];
  const int*   qw    = (const int*)d_in[7];
  float* out = (float*)d_out;

  // Workspace carve-up (bf16 elements). Total ~135 MB.
  __bf16* W    = (__bf16*)d_ws;                        // 18*1024*1024
  __bf16* laB  = W    + (size_t)NLAY * DIM * DIM;      // 18*32*1024
  __bf16* lbB  = laB  + (size_t)NLAY * RANK * DIM;     // 18*1024*32
  __bf16* bufA = lbB  + (size_t)NLAY * DIM * RANK;     // 16384*1024
  __bf16* bufB = bufA + (size_t)BATCH * DIM;
  __bf16* bufC = bufB + (size_t)BATCH * DIM;
  __bf16* T    = bufC + (size_t)BATCH * DIM;           // 16384*32

  dequant_w_kernel<<<(NLAY * DIM * 64 + 255) / 256, 256, 0, stream>>>(qw, scale, W);
  f32_to_bf16_kernel<<<(NLAY * RANK * DIM + 255) / 256, 256, 0, stream>>>(la, laB, NLAY * RANK * DIM);
  f32_to_bf16_kernel<<<(NLAY * DIM * RANK + 255) / 256, 256, 0, stream>>>(lb, lbB, NLAY * DIM * RANK);
  f32_to_bf16_kernel<<<(BATCH * DIM + 255) / 256, 256, 0, stream>>>(x, bufA, BATCH * DIM);

  const dim3 ggrid(BATCH / 128, DIM / 128);   // 128 x 8 blocks
  for (int blk = 0; blk < 6; ++blk) {
    int li = blk * 3;
    lora_t_kernel<<<BATCH / 128, 256, 0, stream>>>(bufA, laB + (size_t)li * RANK * DIM, T);
    qlora_gemm_kernel<<<ggrid, 256, 0, stream>>>(bufA, W + (size_t)li * DIM * DIM,
                                                 bias + li * DIM, T,
                                                 lbB + (size_t)li * DIM * RANK, bufB, 1);
    ++li;
    lora_t_kernel<<<BATCH / 128, 256, 0, stream>>>(bufB, laB + (size_t)li * RANK * DIM, T);
    qlora_gemm_kernel<<<ggrid, 256, 0, stream>>>(bufB, W + (size_t)li * DIM * DIM,
                                                 bias + li * DIM, T,
                                                 lbB + (size_t)li * DIM * RANK, bufC, 1);
    ++li;
    lora_t_kernel<<<BATCH / 128, 256, 0, stream>>>(bufC, laB + (size_t)li * RANK * DIM, T);
    qlora_gemm_kernel<<<ggrid, 256, 0, stream>>>(bufC, W + (size_t)li * DIM * DIM,
                                                 bias + li * DIM, T,
                                                 lbB + (size_t)li * DIM * RANK, bufB, 0);
    if (blk < 5) {
      resid_ln_kernel<<<BATCH, 256, 0, stream>>>(bufB, bufA, ln_w + blk * DIM,
                                                 ln_b + blk * DIM, bufA);
    } else {
      resid_out_kernel<<<(BATCH * DIM) / 256, 256, 0, stream>>>(bufB, bufA, out);
    }
  }
}
